// Text2Motion_Transformer_57062935494867
// MI455X (gfx1250) — matrix operand order repaired and checked by reading, not compiled, not run on hardware
//
#include <hip/hip_runtime.h>
#include <math.h>

// ---------------- problem constants ----------------
#define NLAYER 8
#define BATCH  4
#define T_SEQ  2048
#define C_DIM  1024
#define H_NUM  16
#define HD     64
#define DFF    4096
#define MROWS  (BATCH * T_SEQ)   // 8192

typedef __attribute__((ext_vector_type(16))) __bf16 v16bf;
typedef __attribute__((ext_vector_type(8)))  float  v8f;
typedef __attribute__((ext_vector_type(4)))  unsigned int v4u;
typedef __attribute__((ext_vector_type(8)))  int    v8i;
typedef __attribute__((ext_vector_type(4)))  int    v4i;

#if __has_builtin(__builtin_amdgcn_tensor_load_to_lds)
#define HAVE_TDM 1
#endif

// =====================================================================
// TDM: 2D bf16 tile load, global -> LDS, with LDS row padding.
// =====================================================================
#if defined(HAVE_TDM)
__device__ __forceinline__ void tdm_load_2d_bf16(
    const void* gsrc, unsigned lds_addr,
    unsigned tile_d0, unsigned tile_d1,
    unsigned long long tensor_d0, unsigned long long tensor_d1,
    unsigned long long stride_d0,
    unsigned pad_iv, unsigned pad_am)
{
    const unsigned long long ga = (unsigned long long)(uintptr_t)gsrc;
    v4u g0;
    g0[0] = 1u;                                          // count=1, user desc
    g0[1] = lds_addr;                                    // lds_addr[63:32]
    g0[2] = (unsigned)(ga & 0xffffffffu);                // global_addr[95:64]
    g0[3] = (unsigned)((ga >> 32) & 0x1ffffffu) | (2u << 30);  // [120:96]|type=2

    v8i g1;
    unsigned w0 = (1u << 16)                             // data_size = 2B
                | (1u << 20)                             // pad_enable
                | (pad_iv << 22) | (pad_am << 25);
    g1[0] = (int)w0;
    g1[1] = (int)((unsigned)(tensor_d0 & 0xffffu) << 16);
    g1[2] = (int)(((unsigned)((tensor_d0 >> 16) & 0xffffu)) |
                  ((unsigned)(tensor_d1 & 0xffffu) << 16));
    g1[3] = (int)(((unsigned)((tensor_d1 >> 16) & 0xffffu)) |
                  (tile_d0 << 16));
    g1[4] = (int)(tile_d1 & 0xffffu);                    // tile_dim1, tile_dim2=0
    g1[5] = (int)(unsigned)(stride_d0 & 0xffffffffu);    // dim0_stride[31:0]
    g1[6] = (int)(unsigned)((stride_d0 >> 32) & 0xffffu);
    g1[7] = 0;

    v4i g2 = {0, 0, 0, 0}, g3 = {0, 0, 0, 0};
#if __clang_major__ >= 23
    v8i g1b = {0, 0, 0, 0, 0, 0, 0, 0};
    __builtin_amdgcn_tensor_load_to_lds(g0, g1, g2, g3, g1b, 0);
#else
    __builtin_amdgcn_tensor_load_to_lds(g0, g1, g2, g3, 0);
#endif
}
#define TDM_WAIT(n) \
    do { if ((threadIdx.x >> 5) == 0) __builtin_amdgcn_s_wait_tensorcnt(n); } while (0)
#else
#define TDM_WAIT(n) do { } while (0)
#endif

#define LDS_OFF(p) ((unsigned)(uintptr_t)(p))

// =====================================================================
// LayerNorm (fp32 in) -> bf16 out.  One block per row, 256 threads.
// =====================================================================
__global__ __launch_bounds__(256)
void layernorm_bf16(const float* __restrict__ x, const float* __restrict__ g,
                    const float* __restrict__ bta, __bf16* __restrict__ out)
{
    __shared__ float red[8];
    const int row = blockIdx.x;
    const int tid = threadIdx.x;
    const int wid = tid >> 5;
    const float* xr = x + (size_t)row * C_DIM;

    float v4[4];
    float s = 0.f;
#pragma unroll
    for (int i = 0; i < 4; ++i) { v4[i] = xr[tid + i * 256]; s += v4[i]; }
#pragma unroll
    for (int off = 16; off > 0; off >>= 1) s += __shfl_xor(s, off, 32);
    if ((tid & 31) == 0) red[wid] = s;
    __syncthreads();
    float tot = 0.f;
#pragma unroll
    for (int i = 0; i < 8; ++i) tot += red[i];
    const float mu = tot * (1.0f / C_DIM);
    __syncthreads();

    float s2 = 0.f;
#pragma unroll
    for (int i = 0; i < 4; ++i) { float d = v4[i] - mu; s2 += d * d; }
#pragma unroll
    for (int off = 16; off > 0; off >>= 1) s2 += __shfl_xor(s2, off, 32);
    if ((tid & 31) == 0) red[wid] = s2;
    __syncthreads();
    float tot2 = 0.f;
#pragma unroll
    for (int i = 0; i < 8; ++i) tot2 += red[i];
    const float inv = rsqrtf(tot2 * (1.0f / C_DIM) + 1e-5f);

    __bf16* orow = out + (size_t)row * C_DIM;
#pragma unroll
    for (int i = 0; i < 4; ++i) {
        const int c = tid + i * 256;
        orow[c] = (__bf16)((v4[i] - mu) * inv * g[c] + bta[c]);
    }
}

// =====================================================================
// GEMM:  out[M,N] = A[M,K](bf16) x W[K,N](fp32->bf16) + bias
// MODE 0: bf16 = result   MODE 1: bf16 = gelu(result)
// MODE 2: f32  = resid + result (in-place ok)
// 256 threads / 8 waves; block tile 128x128; wave tile 32x64; K-step 32.
// A staged by TDM (double-buffered); W converted+TRANSPOSED via VALU so
// both A and B fragments gather as contiguous ds_load_b128.
// =====================================================================
template <int MODE>
__global__ __launch_bounds__(256)
void gemm_wmma(const __bf16* __restrict__ A, const float* __restrict__ W,
               const float* __restrict__ bias, int K, int N,
               const float* __restrict__ resid,
               __bf16* __restrict__ outb, float* __restrict__ outf)
{
    __shared__ __bf16 Als[2][128][40];   // [m][k]: 64B data + 16B pad
    __shared__ __bf16 Wt[2][128][40];    // [n][k]: transposed weight tile

    const int tid  = threadIdx.x;
    const int lane = tid & 31;
    const int wid  = tid >> 5;
    const int lrow = lane & 15;
    const int lhi  = lane >> 4;
    const int wm   = (wid & 3) * 32;
    const int wn   = (wid >> 2) * 64;
    const int m0   = blockIdx.y * 128;
    const int n0   = blockIdx.x * 128;

    v8f acc[2][4];
#pragma unroll
    for (int i = 0; i < 2; ++i)
#pragma unroll
        for (int j = 0; j < 4; ++j)
            acc[i][j] = (v8f){0.f, 0.f, 0.f, 0.f, 0.f, 0.f, 0.f, 0.f};

    const int arow = tid >> 1, acol = (tid & 1) * 16;  // fallback A mapping
    const int wrow = tid >> 3, wcol = (tid & 7) * 16;  // W: k = wrow, n = wcol..+15

    auto stageA = [&](int buf, int k0) {
#if defined(HAVE_TDM)
        if (wid == 0)
            tdm_load_2d_bf16(A + (size_t)m0 * K + k0, LDS_OFF(&Als[buf][0][0]),
                             32, 128,
                             (unsigned long long)K, (unsigned long long)MROWS,
                             (unsigned long long)K,
                             /*pad_iv: 16dw=64B*/3, /*pad_am: 4dw=16B*/3);
#else
        const __bf16* Ap = A + (size_t)(m0 + arow) * K + k0 + acol;
#pragma unroll
        for (int i = 0; i < 16; ++i) Als[buf][arow][acol + i] = Ap[i];
#endif
    };
    auto stageW = [&](int buf, int k0) {   // read row-major, store transposed
        const float* Wp = W + (size_t)(k0 + wrow) * N + n0 + wcol;
#pragma unroll
        for (int i = 0; i < 16; ++i) Wt[buf][wcol + i][wrow] = (__bf16)Wp[i];
    };

    const int KSTEPS = K >> 5;
    stageA(0, 0);
    stageW(0, 0);

    for (int s = 0; s < KSTEPS; ++s) {
        const int k0  = s << 5;
        const int cur = s & 1, nxt = cur ^ 1;
        const bool more = (s + 1) < KSTEPS;
        if (more) {
            stageA(nxt, k0 + 32);
            stageW(nxt, k0 + 32);
            __builtin_prefetch(W + (size_t)(k0 + 64 + wrow) * N + n0 + wcol, 0, 1);
        }
        if (more) TDM_WAIT(1); else TDM_WAIT(0);
        __syncthreads();

        v16bf af[2], bf[4];
#pragma unroll
        for (int tm = 0; tm < 2; ++tm)
#pragma unroll
            for (int e = 0; e < 16; ++e)
                af[tm][e] = Als[cur][wm + 16 * tm + lrow][lhi * 16 + e];
#pragma unroll
        for (int tn = 0; tn < 4; ++tn)
#pragma unroll
            for (int e = 0; e < 16; ++e)
                bf[tn][e] = Wt[cur][wn + 16 * tn + lrow][lhi * 16 + e];

#pragma unroll
        for (int tm = 0; tm < 2; ++tm)
#pragma unroll
            for (int tn = 0; tn < 4; ++tn)
                acc[tm][tn] = __builtin_amdgcn_wmma_f32_16x16x32_bf16(
                    false, af[tm], false, bf[tn], (short)0, acc[tm][tn], false, false);
        __syncthreads();
    }

#pragma unroll
    for (int tm = 0; tm < 2; ++tm)
#pragma unroll
        for (int tn = 0; tn < 4; ++tn)
#pragma unroll
            for (int e8 = 0; e8 < 8; ++e8) {
                const int R    = m0 + wm + 16 * tm + e8 + 8 * lhi;
                const int Ncol = n0 + wn + 16 * tn + lrow;
                const float val = acc[tm][tn][e8] + bias[Ncol];
                const size_t idx = (size_t)R * N + Ncol;
                if (MODE == 0) {
                    outb[idx] = (__bf16)val;
                } else if (MODE == 1) {
                    outb[idx] = (__bf16)(0.5f * val * (1.0f + erff(val * 0.70710678118f)));
                } else {
                    outf[idx] = resid[idx] + val;
                }
            }
}

// =====================================================================
// Flash attention (causal).  Grid: B*H*(T/64) blocks of 128 threads.
// K tiles staged by TDM (double-buffered); V staged VALU-transposed so
// P*V B-fragments vectorize.  Diagonal tile peeled (mask-free main loop);
// softmax scale folded into Q (0.125 exact in bf16).
// =====================================================================
__global__ __launch_bounds__(128)
void flash_attn(const __bf16* __restrict__ q, const __bf16* __restrict__ k,
                const __bf16* __restrict__ v, __bf16* __restrict__ y)
{
    __shared__ __bf16 Kls[2][64][72];   // [key][d]
    __shared__ __bf16 Vt [2][64][72];   // [d][key]  (transposed)
    __shared__ __bf16 Pls[4][16][72];

    const int NQ  = T_SEQ / 64;
    const int iq  = blockIdx.x % NQ;
    const int bh  = blockIdx.x / NQ;
    const int b   = bh / H_NUM;
    const int h   = bh % H_NUM;
    const int tid = threadIdx.x;
    const int lane = tid & 31, wq = tid >> 5, wid = wq;
    const int lrow = lane & 15, lhi = lane >> 4;
    const int qbase = iq * 64 + wq * 16;

    const int ldr = tid >> 1, ldc = (tid & 1) * 32;

    auto stageKV = [&](int buf, int j) {
        const size_t base = ((size_t)(b * T_SEQ + j * 64)) * C_DIM + h * HD;
#if defined(HAVE_TDM)
        if (wid == 0)
            tdm_load_2d_bf16(k + base, LDS_OFF(&Kls[buf][0][0]),
                             64, 64, (unsigned long long)C_DIM,
                             (unsigned long long)MROWS,
                             (unsigned long long)C_DIM,
                             /*pad_iv: 32dw=128B*/4, /*pad_am: 4dw=16B*/3);
#else
        {
            const __bf16* kp = k + base + (size_t)ldr * C_DIM + ldc;
#pragma unroll
            for (int i = 0; i < 32; ++i) Kls[buf][ldr][ldc + i] = kp[i];
        }
#endif
        const __bf16* vp = v + base + (size_t)ldr * C_DIM + ldc;
#pragma unroll
        for (int i = 0; i < 32; ++i) Vt[buf][ldc + i][ldr] = vp[i];  // transpose
    };

    // Q fragments with softmax scale pre-folded (1/sqrt(64) exact in bf16)
    v16bf qa[2];
#pragma unroll
    for (int dc = 0; dc < 2; ++dc)
#pragma unroll
        for (int e = 0; e < 16; ++e) {
            const int d = dc * 32 + lhi * 16 + e;
            const float qv =
                (float)q[((size_t)(b * T_SEQ + qbase + lrow)) * C_DIM + h * HD + d];
            qa[dc][e] = (__bf16)(qv * 0.125f);
        }

    v8f O[4], mprev, lsum;
#pragma unroll
    for (int dn = 0; dn < 4; ++dn)
        O[dn] = (v8f){0.f, 0.f, 0.f, 0.f, 0.f, 0.f, 0.f, 0.f};
#pragma unroll
    for (int e8 = 0; e8 < 8; ++e8) { mprev[e8] = -INFINITY; lsum[e8] = 0.f; }

    auto tile = [&](int cur, int j, bool domask) {
        // S = (Q/8) K^T   (4 n-tiles of 16 keys)
        v8f S[4];
#pragma unroll
        for (int nt = 0; nt < 4; ++nt) {
            S[nt] = (v8f){0.f, 0.f, 0.f, 0.f, 0.f, 0.f, 0.f, 0.f};
#pragma unroll
            for (int dc = 0; dc < 2; ++dc) {
                v16bf kb;
#pragma unroll
                for (int e = 0; e < 16; ++e)
                    kb[e] = Kls[cur][nt * 16 + lrow][dc * 32 + lhi * 16 + e];
                S[nt] = __builtin_amdgcn_wmma_f32_16x16x32_bf16(
                    false, qa[dc], false, kb, (short)0, S[nt], false, false);
            }
            if (domask) {
#pragma unroll
                for (int e8 = 0; e8 < 8; ++e8) {
                    const int key = j * 64 + nt * 16 + lrow;
                    const int qr  = qbase + e8 + 8 * lhi;
                    if (key > qr) S[nt][e8] = -INFINITY;
                }
            }
        }

        // online softmax (row = e8 + 8*lhi; cols across 16 lanes of the half)
        v8f mcur;
#pragma unroll
        for (int e8 = 0; e8 < 8; ++e8) {
            float m = S[0][e8];
#pragma unroll
            for (int nt = 1; nt < 4; ++nt) m = fmaxf(m, S[nt][e8]);
            mcur[e8] = m;
        }
#pragma unroll
        for (int off = 8; off > 0; off >>= 1)
#pragma unroll
            for (int e8 = 0; e8 < 8; ++e8)
                mcur[e8] = fmaxf(mcur[e8], __shfl_xor(mcur[e8], off, 32));

        v8f mnew, alpha, rs;
#pragma unroll
        for (int e8 = 0; e8 < 8; ++e8) {
            mnew[e8]  = fmaxf(mprev[e8], mcur[e8]);
            alpha[e8] = expf(mprev[e8] - mnew[e8]);
            rs[e8]    = 0.f;
        }
#pragma unroll
        for (int nt = 0; nt < 4; ++nt)
#pragma unroll
            for (int e8 = 0; e8 < 8; ++e8) {
                const float p = expf(S[nt][e8] - mnew[e8]);
                rs[e8] += p;
                Pls[wq][e8 + 8 * lhi][nt * 16 + lrow] = (__bf16)p;
            }
#pragma unroll
        for (int off = 8; off > 0; off >>= 1)
#pragma unroll
            for (int e8 = 0; e8 < 8; ++e8)
                rs[e8] += __shfl_xor(rs[e8], off, 32);
#pragma unroll
        for (int e8 = 0; e8 < 8; ++e8) {
            lsum[e8]  = lsum[e8] * alpha[e8] + rs[e8];
            mprev[e8] = mnew[e8];
        }
#pragma unroll
        for (int dn = 0; dn < 4; ++dn)
#pragma unroll
            for (int e8 = 0; e8 < 8; ++e8) O[dn][e8] *= alpha[e8];

        // Pls is wave-private: wave-local LDS drain instead of block barrier
        asm volatile("s_wait_dscnt 0x0" ::: "memory");
        __builtin_amdgcn_wave_barrier();

        // O += P V   (4 d-tiles of 16, key-chunks of 32)
#pragma unroll
        for (int dn = 0; dn < 4; ++dn)
#pragma unroll
            for (int kc = 0; kc < 2; ++kc) {
                v16bf pa, vb;
#pragma unroll
                for (int e = 0; e < 16; ++e)
                    pa[e] = Pls[wq][lrow][kc * 32 + lhi * 16 + e];
#pragma unroll
                for (int e = 0; e < 16; ++e)
                    vb[e] = Vt[cur][dn * 16 + lrow][kc * 32 + lhi * 16 + e];
                O[dn] = __builtin_amdgcn_wmma_f32_16x16x32_bf16(
                    false, pa, false, vb, (short)0, O[dn], false, false);
            }
    };

    stageKV(0, 0);

    for (int j = 0; j < iq; ++j) {          // mask-free steady state
        const int cur = j & 1;
        stageKV(cur ^ 1, j + 1);
        TDM_WAIT(1);
        __syncthreads();
        tile(cur, j, false);
        __syncthreads();
    }
    {                                        // peeled diagonal tile
        const int cur = iq & 1;
        TDM_WAIT(0);
        __syncthreads();
        tile(cur, iq, true);
    }

#pragma unroll
    for (int dn = 0; dn < 4; ++dn)
#pragma unroll
        for (int e8 = 0; e8 < 8; ++e8) {
            const int qr = qbase + e8 + 8 * lhi;
            const float o = O[dn][e8] / lsum[e8];
            y[((size_t)(b * T_SEQ + qr)) * C_DIM + h * HD + dn * 16 + lrow] = (__bf16)o;
        }
}

// =====================================================================
// Host orchestration
// =====================================================================
extern "C" void kernel_launch(void* const* d_in, const int* in_sizes, int n_in,
                              void* d_out, int out_size, void* d_ws, size_t ws_size,
                              hipStream_t stream)
{
    (void)in_sizes; (void)n_in; (void)out_size; (void)ws_size;

    const float* x    = (const float*)d_in[0];
    const float* ln1g = (const float*)d_in[1];
    const float* ln1b = (const float*)d_in[2];
    const float* ln2g = (const float*)d_in[3];
    const float* ln2b = (const float*)d_in[4];
    const float* wq   = (const float*)d_in[5];
    const float* bq   = (const float*)d_in[6];
    const float* wk   = (const float*)d_in[7];
    const float* bk   = (const float*)d_in[8];
    const float* wv   = (const float*)d_in[9];
    const float* bv   = (const float*)d_in[10];
    const float* wo   = (const float*)d_in[11];
    const float* bo   = (const float*)d_in[12];
    const float* w1   = (const float*)d_in[13];
    const float* b1   = (const float*)d_in[14];
    const float* w2   = (const float*)d_in[15];
    const float* b2   = (const float*)d_in[16];

    float* resid = (float*)d_out;  // residual stream lives in d_out

    char* ws = (char*)d_ws;
    size_t off = 0;
    auto wsalloc = [&](size_t bytes) -> void* {
        void* p = ws + off;
        off += (bytes + 255) & ~(size_t)255;
        return p;
    };
    const size_t actb = (size_t)MROWS * C_DIM * sizeof(__bf16);
    __bf16* hn   = (__bf16*)wsalloc(actb);
    __bf16* qbuf = (__bf16*)wsalloc(actb);
    __bf16* kbuf = (__bf16*)wsalloc(actb);
    __bf16* vbuf = (__bf16*)wsalloc(actb);
    __bf16* ybuf = (__bf16*)wsalloc(actb);
    __bf16* gmid = (__bf16*)wsalloc((size_t)MROWS * DFF * sizeof(__bf16));

    hipMemcpyAsync(resid, x, (size_t)MROWS * C_DIM * sizeof(float),
                   hipMemcpyDeviceToDevice, stream);

    const dim3 blk(256);
    const dim3 gC(C_DIM / 128, MROWS / 128);   // N = 1024
    const dim3 gF(DFF / 128, MROWS / 128);     // N = 4096
    const int attnBlocks = BATCH * H_NUM * (T_SEQ / 64);

    for (int l = 0; l < NLAYER; ++l) {
        const size_t wOff = (size_t)l * C_DIM * C_DIM;
        const size_t fOff = (size_t)l * C_DIM * DFF;
        const size_t cOff = (size_t)l * C_DIM;
        const size_t dOff = (size_t)l * DFF;

        layernorm_bf16<<<MROWS, blk, 0, stream>>>(resid, ln1g + cOff, ln1b + cOff, hn);

        gemm_wmma<0><<<gC, blk, 0, stream>>>(hn, wq + wOff, bq + cOff, C_DIM, C_DIM,
                                             nullptr, qbuf, nullptr);
        gemm_wmma<0><<<gC, blk, 0, stream>>>(hn, wk + wOff, bk + cOff, C_DIM, C_DIM,
                                             nullptr, kbuf, nullptr);
        gemm_wmma<0><<<gC, blk, 0, stream>>>(hn, wv + wOff, bv + cOff, C_DIM, C_DIM,
                                             nullptr, vbuf, nullptr);

        flash_attn<<<attnBlocks, dim3(128), 0, stream>>>(qbuf, kbuf, vbuf, ybuf);

        gemm_wmma<2><<<gC, blk, 0, stream>>>(ybuf, wo + wOff, bo + cOff, C_DIM, C_DIM,
                                             resid, nullptr, resid);

        layernorm_bf16<<<MROWS, blk, 0, stream>>>(resid, ln2g + cOff, ln2b + cOff, hn);

        gemm_wmma<1><<<gF, blk, 0, stream>>>(hn, w1 + fOff, b1 + dOff, C_DIM, DFF,
                                             nullptr, gmid, nullptr);
        gemm_wmma<2><<<gC, blk, 0, stream>>>(gmid, w2 + fOff, b2 + cOff, DFF, C_DIM,
                                             resid, nullptr, resid);
    }
}
